// MIDIGPT_38835094290989
// MI455X (gfx1250) — matrix operand order repaired and checked by reading, not compiled, not run on hardware
//
#include <hip/hip_runtime.h>

typedef __attribute__((ext_vector_type(16))) _Float16 v16h;
typedef __attribute__((ext_vector_type(8)))  _Float16 v8h;
typedef __attribute__((ext_vector_type(8)))  float    v8f;

#define LAYERS 6
#define NHEAD  12
#define EMB    768
#define E4     3072
#define HDIM   64
#define SEQ    1024
#define BATCH  8
#define VOC    512
#define N3     (3*EMB)          // 2304
#define TOK    (BATCH*SEQ)      // 8192

// ---------------------------------------------------------------------------
// WMMA helper (gfx1250, wave32). 16x16x32 f16 -> f32 accumulate.
// ---------------------------------------------------------------------------
__device__ inline v8f wmma32(v16h a, v16h b, v8f c) {
  return __builtin_amdgcn_wmma_f32_16x16x32_f16(
      /*neg_a=*/false, a, /*neg_b=*/false, b,
      /*c_mod=*/(short)0, c, /*reuse_a=*/false, /*reuse_b=*/false);
}

// A-fragment 16x32 f16, element (m,k) at p[m*rs + k], rows 16B-aligned.
// lanes 0-15: M=lane, K {0..7,16..23}; lanes 16-31: M=lane-16, K {8..15,24..31}
__device__ inline v16h load_a16_v(const _Float16* __restrict__ p, int rs) {
  const int lane = threadIdx.x & 31;
  const int m    = lane & 15;
  const int kb   = (lane >> 4) * 8;
  const _Float16* row = p + (size_t)m * rs + kb;
  v8h lo = *(const v8h*)(row);
  v8h hi = *(const v8h*)(row + 16);
  return __builtin_shufflevector(lo, hi, 0,1,2,3,4,5,6,7,8,9,10,11,12,13,14,15);
}

// B-fragment 32x16 f16 from K-innermost storage: element (kk,n) at p[n*ns+kk].
// lanes 0-15: N=lane, K {0..7,16..23}; lanes 16-31: N=lane-16, K {8..15,24..31}.
__device__ inline v16h load_b16_v(const _Float16* __restrict__ p, int ns) {
  const int lane = threadIdx.x & 31;
  const int n    = lane & 15;
  const int kb   = (lane >> 4) * 8;
  const _Float16* q = p + (size_t)n * ns + kb;
  v8h lo = *(const v8h*)(q);
  v8h hi = *(const v8h*)(q + 16);
  return __builtin_shufflevector(lo, hi, 0,1,2,3,4,5,6,7,8,9,10,11,12,13,14,15);
}

// ---------------------------------------------------------------------------
// Tiled GEMM:  C[M,N] = act(A[M,K](f16,row-major) * Bt[N,K](f16,K-innermost)
//                           + bias)
// block = 128 threads (4 waves), 64x64 tile.  K staged in 32-slabs via LDS,
// software-pipelined through registers: next slab's global_load_b128s issue
// before the current slab's WMMAs, waited only at the following ds_store.
// ---------------------------------------------------------------------------
__global__ __launch_bounds__(128)
void gemm_wmma_f16(const _Float16* __restrict__ A, const _Float16* __restrict__ Bt,
                   float* __restrict__ Cf, _Float16* __restrict__ Ch,
                   const float* __restrict__ bias, int M, int N, int K, int act) {
  __shared__ __align__(16) _Float16 As[64 * 40];
  __shared__ __align__(16) _Float16 Bs[64 * 40];
  const int tid  = threadIdx.x;
  const int lane = tid & 31;
  const int w    = tid >> 5;
  const int row0 = blockIdx.y * 64;
  const int col0 = blockIdx.x * 64;

  // staging assignment: thread handles rows r0 and r0+32, 8-half column co
  const int r0 = tid >> 2;
  const int co = (tid & 3) * 8;
  const _Float16* pa0 = A  + (size_t)(row0 + r0)      * K + co;
  const _Float16* pa1 = A  + (size_t)(row0 + r0 + 32) * K + co;
  const _Float16* pb0 = Bt + (size_t)(col0 + r0)      * K + co;
  const _Float16* pb1 = Bt + (size_t)(col0 + r0 + 32) * K + co;
  _Float16* sa0 = &As[r0 * 40 + co];
  _Float16* sa1 = &As[(r0 + 32) * 40 + co];
  _Float16* sb0 = &Bs[r0 * 40 + co];
  _Float16* sb1 = &Bs[(r0 + 32) * 40 + co];

  // preload slab 0
  v8h ra0 = *(const v8h*)pa0;
  v8h ra1 = *(const v8h*)pa1;
  v8h rb0 = *(const v8h*)pb0;
  v8h rb1 = *(const v8h*)pb1;

  v8f acc[4];
  v8f zero = {};
#pragma unroll
  for (int i = 0; i < 4; ++i) acc[i] = zero;

  for (int k0 = 0; k0 < K; k0 += 32) {
    // commit staged slab to LDS (waits on the in-flight loads here)
    *(v8h*)sa0 = ra0;
    *(v8h*)sa1 = ra1;
    *(v8h*)sb0 = rb0;
    *(v8h*)sb1 = rb1;
    __syncthreads();
    // issue next slab's loads; they overlap with the WMMAs below
    if (k0 + 32 < K) {
      pa0 += 32; pa1 += 32; pb0 += 32; pb1 += 32;
      ra0 = *(const v8h*)pa0;
      ra1 = *(const v8h*)pa1;
      rb0 = *(const v8h*)pb0;
      rb1 = *(const v8h*)pb1;
    }
    v16h af = load_a16_v(&As[w * 16 * 40], 40);
#pragma unroll
    for (int nt = 0; nt < 4; ++nt) {
      v16h bf = load_b16_v(&Bs[nt * 16 * 40], 40);
      acc[nt] = wmma32(af, bf, acc[nt]);
    }
    __syncthreads();
  }

  const int roff = (lane >> 4) * 8;
  const int ncol = lane & 15;
#pragma unroll
  for (int nt = 0; nt < 4; ++nt) {
    const int col = col0 + nt * 16 + ncol;
    const float bv = bias ? bias[col] : 0.0f;
#pragma unroll
    for (int r = 0; r < 8; ++r) {
      const int row = row0 + w * 16 + roff + r;
      float v = acc[nt][r] + bv;
      if (act) v = 0.5f * v * (1.0f + erff(v * 0.70710678118654752f));
      const size_t idx = (size_t)row * N + col;
      if (Cf) Cf[idx] = v;
      if (Ch) Ch[idx] = (_Float16)v;
    }
  }
}

// ---------------------------------------------------------------------------
// Flash attention.  qkv: [TOK, 3E] (q|k|v, head h at +h*64).
// vt:  [B,H,HDIM,SEQ] f16 (seq-innermost V for vectorized P*V fragments).
// out: [TOK, E] f16.  One block = 64 query rows of one (b,h); 4 waves x 16.
// ---------------------------------------------------------------------------
__global__ __launch_bounds__(128)
void attn_flash(const _Float16* __restrict__ qkv, const _Float16* __restrict__ vt,
                _Float16* __restrict__ out) {
  __shared__ __align__(16) _Float16 probs[4][16 * 40];
  const int tid  = threadIdx.x;
  const int lane = tid & 31;
  const int w    = tid >> 5;
  const int bh   = blockIdx.y;
  const int b    = bh / NHEAD, h = bh % NHEAD;
  const int qb   = blockIdx.x * 64 + w * 16;

  const size_t rowbase = (size_t)b * SEQ * N3;
  const _Float16* qp = qkv + rowbase + h * HDIM;
  const _Float16* kp = qkv + rowbase + EMB + h * HDIM;
  const _Float16* vtp = vt + (size_t)bh * HDIM * SEQ;

  v16h aq0 = load_a16_v(qp + (size_t)qb * N3 + 0,  N3);
  v16h aq1 = load_a16_v(qp + (size_t)qb * N3 + 32, N3);

  v8f o[4];
  v8f zero = {};
#pragma unroll
  for (int i = 0; i < 4; ++i) o[i] = zero;
  float mrun[8], lsum[8];
#pragma unroll
  for (int r = 0; r < 8; ++r) { mrun[r] = -3.0e38f; lsum[r] = 0.0f; }

  const int roff   = (lane >> 4) * 8;
  const int ncol   = lane & 15;
  const int ntiles = (qb + 15) / 32 + 1;
  const float scale = 0.125f;  // HD^-0.5

  for (int kt = 0; kt < ntiles; ++kt) {
    const int kb = kt * 32;
    if (kt + 1 < ntiles) {  // prefetch next key/value tile
      __builtin_prefetch(kp + (size_t)(kb + 32 + lane) * N3, 0, 1);
      __builtin_prefetch(vtp + (size_t)(lane * 2) * SEQ + kb + 32, 0, 1);
    }
    v8f s0 = zero, s1 = zero;
    {
      v16h b00 = load_b16_v(kp + (size_t)kb * N3 + 0,  N3);
      v16h b01 = load_b16_v(kp + (size_t)kb * N3 + 32, N3);
      s0 = wmma32(aq0, b00, s0);
      s0 = wmma32(aq1, b01, s0);
      v16h b10 = load_b16_v(kp + (size_t)(kb + 16) * N3 + 0,  N3);
      v16h b11 = load_b16_v(kp + (size_t)(kb + 16) * N3 + 32, N3);
      s1 = wmma32(aq0, b10, s1);
      s1 = wmma32(aq1, b11, s1);
    }
#pragma unroll
    for (int r = 0; r < 8; ++r) {
      const int qrow = qb + roff + r;
      float x0 = s0[r] * scale; if (kb + ncol      > qrow) x0 = -3.0e38f;
      float x1 = s1[r] * scale; if (kb + 16 + ncol > qrow) x1 = -3.0e38f;
      float mx = fmaxf(x0, x1);
      for (int off = 1; off < 16; off <<= 1)
        mx = fmaxf(mx, __shfl_xor(mx, off, 32));
      const float mnew  = fmaxf(mrun[r], mx);
      const float alpha = __expf(mrun[r] - mnew);
      const float p0 = __expf(x0 - mnew);
      const float p1 = __expf(x1 - mnew);
      float rs = p0 + p1;
      for (int off = 1; off < 16; off <<= 1)
        rs += __shfl_xor(rs, off, 32);
      lsum[r] = lsum[r] * alpha + rs;
      mrun[r] = mnew;
#pragma unroll
      for (int i = 0; i < 4; ++i) o[i][r] *= alpha;
      probs[w][(roff + r) * 40 + ncol]      = (_Float16)p0;
      probs[w][(roff + r) * 40 + 16 + ncol] = (_Float16)p1;
    }
    asm volatile("s_wait_dscnt 0" ::: "memory");
    __builtin_amdgcn_wave_barrier();
    v16h ap = load_a16_v(&probs[w][0], 40);
#pragma unroll
    for (int hs = 0; hs < 4; ++hs) {
      v16h bv = load_b16_v(vtp + (size_t)(hs * 16) * SEQ + kb, SEQ);
      o[hs] = wmma32(ap, bv, o[hs]);
    }
    __builtin_amdgcn_wave_barrier();
  }

  _Float16* op = out + (size_t)(b * SEQ + qb) * EMB + h * HDIM;
#pragma unroll
  for (int r = 0; r < 8; ++r) {
    const float inv = 1.0f / lsum[r];
    const int row = roff + r;
#pragma unroll
    for (int hs = 0; hs < 4; ++hs)
      op[(size_t)row * EMB + hs * 16 + ncol] = (_Float16)(o[hs][r] * inv);
  }
}

// ---------------------------------------------------------------------------
// LayerNorm (+ optional residual): one block per token row, E=768=3*256.
// ---------------------------------------------------------------------------
__global__ __launch_bounds__(256)
void ln_kernel(const float* __restrict__ inp, const float* __restrict__ resid,
               const float* __restrict__ g, const float* __restrict__ bta,
               float* __restrict__ of, _Float16* __restrict__ oh) {
  __shared__ float red[256];
  const int t = blockIdx.x, tid = threadIdx.x;
  const size_t base = (size_t)t * EMB;
  float vals[3];
  float s = 0.0f;
#pragma unroll
  for (int i = 0; i < 3; ++i) {
    const int e = tid + i * 256;
    float x = inp[base + e];
    if (resid) x += resid[base + e];
    vals[i] = x; s += x;
  }
  red[tid] = s; __syncthreads();
  for (int st = 128; st > 0; st >>= 1) {
    if (tid < st) red[tid] += red[tid + st];
    __syncthreads();
  }
  const float mean = red[0] * (1.0f / EMB);
  __syncthreads();
  float v = 0.0f;
#pragma unroll
  for (int i = 0; i < 3; ++i) { const float d = vals[i] - mean; v += d * d; }
  red[tid] = v; __syncthreads();
  for (int st = 128; st > 0; st >>= 1) {
    if (tid < st) red[tid] += red[tid + st];
    __syncthreads();
  }
  const float rstd = rsqrtf(red[0] * (1.0f / EMB) + 1e-5f);
#pragma unroll
  for (int i = 0; i < 3; ++i) {
    const int e = tid + i * 256;
    const float y = (vals[i] - mean) * rstd * g[e] + bta[e];
    if (of) of[base + e] = y;
    oh[base + e] = (_Float16)y;
  }
}

// ---------------------------------------------------------------------------
// Embedding: x = tok_emb[idx] + pos_emb[s]; emit f32 + f16.
// ---------------------------------------------------------------------------
__global__ void embed_kernel(const int* __restrict__ idx,
                             const float* __restrict__ tok,
                             const float* __restrict__ pos,
                             float* __restrict__ xf, _Float16* __restrict__ xh) {
  const long long total = (long long)TOK * EMB;
  for (long long i = (long long)blockIdx.x * blockDim.x + threadIdx.x; i < total;
       i += (long long)gridDim.x * blockDim.x) {
    const int e = (int)(i % EMB);
    const int t = (int)(i / EMB);
    const int s = t % SEQ;
    const float v = tok[(size_t)idx[t] * EMB + e] + pos[(size_t)s * EMB + e];
    xf[i] = v; xh[i] = (_Float16)v;
  }
}

// f32 [L,K,N] -> f16 transposed [L,N,K]  (K-innermost for B fragments)
__global__ void cvt_t_kernel(const float* __restrict__ src,
                             _Float16* __restrict__ dst,
                             int Lc, int K, int N) {
  const long long total = (long long)Lc * K * N;
  const long long perL = (long long)K * N;
  for (long long i = (long long)blockIdx.x * blockDim.x + threadIdx.x; i < total;
       i += (long long)gridDim.x * blockDim.x) {
    const int l = (int)(i / perL);
    const long long rem = i % perL;
    const int n = (int)(rem / K);
    const int k = (int)(rem % K);
    dst[i] = (_Float16)src[((size_t)l * K + k) * N + n];
  }
}

// Repack Wq/Wk/Wv [L,H,E,HD] -> fused f16 [L][3E][E]  (E=K innermost)
__global__ void pack_qkv_kernel(const float* __restrict__ Wq,
                                const float* __restrict__ Wk,
                                const float* __restrict__ Wv,
                                _Float16* __restrict__ dst) {
  const long long total = (long long)LAYERS * N3 * EMB;
  for (long long i = (long long)blockIdx.x * blockDim.x + threadIdx.x; i < total;
       i += (long long)gridDim.x * blockDim.x) {
    long long rem = i;
    const int e   = (int)(rem % EMB); rem /= EMB;
    const int col = (int)(rem % N3);
    const int l   = (int)(rem / N3);
    const int sel = col / EMB, c2 = col % EMB;
    const int h = c2 / HDIM, d = c2 % HDIM;
    const float* src = (sel == 0) ? Wq : (sel == 1) ? Wk : Wv;
    dst[i] = (_Float16)src[(((size_t)l * NHEAD + h) * EMB + e) * HDIM + d];
  }
}

// Repack V slice of qkv [TOK,3E] -> VT [B,H,HDIM,SEQ]  (seq-innermost)
__global__ void repack_vt_kernel(const _Float16* __restrict__ qkv,
                                 _Float16* __restrict__ vt) {
  const long long total = (long long)TOK * EMB;
  for (long long i = (long long)blockIdx.x * blockDim.x + threadIdx.x; i < total;
       i += (long long)gridDim.x * blockDim.x) {
    const int s = (int)(i % SEQ);
    const int c = (int)(i / SEQ);          // (b*H + h)*HDIM + d
    const int d = c % HDIM;
    const int bh = c / HDIM;
    const int h = bh % NHEAD, b = bh / NHEAD;
    vt[i] = qkv[(size_t)(b * SEQ + s) * N3 + 2 * EMB + h * HDIM + d];
  }
}

// ---------------------------------------------------------------------------
extern "C" void kernel_launch(void* const* d_in, const int* in_sizes, int n_in,
                              void* d_out, int out_size, void* d_ws, size_t ws_size,
                              hipStream_t stream) {
  const int*   indices = (const int*)d_in[0];
  const float* tok     = (const float*)d_in[1];
  const float* pos     = (const float*)d_in[2];
  const float* Wq      = (const float*)d_in[3];
  const float* Wk      = (const float*)d_in[4];
  const float* Wv      = (const float*)d_in[5];
  const float* Wo      = (const float*)d_in[6];
  const float* bo      = (const float*)d_in[7];
  const float* ln1g    = (const float*)d_in[8];
  const float* ln1b    = (const float*)d_in[9];
  const float* ln2g    = (const float*)d_in[10];
  const float* ln2b    = (const float*)d_in[11];
  const float* W1      = (const float*)d_in[12];
  const float* b1      = (const float*)d_in[13];
  const float* W2      = (const float*)d_in[14];
  const float* b2      = (const float*)d_in[15];
  const float* lnfg    = (const float*)d_in[16];
  const float* lnfb    = (const float*)d_in[17];
  const float* Whead   = (const float*)d_in[18];
  const float* bhead   = (const float*)d_in[19];

  // -------- workspace carve (256B aligned) --------
  char* wp = (char*)d_ws;
  auto carve = [&](size_t bytes) -> void* {
    void* p = wp; wp += (bytes + 255) & ~(size_t)255; return p;
  };
  _Float16* WQKV  = (_Float16*)carve((size_t)LAYERS * N3 * EMB * 2);   // [L][3E][E]
  _Float16* WOt   = (_Float16*)carve((size_t)LAYERS * EMB * EMB * 2);  // [L][E][E]
  _Float16* W1t   = (_Float16*)carve((size_t)LAYERS * E4 * EMB * 2);   // [L][4E][E]
  _Float16* W2t   = (_Float16*)carve((size_t)LAYERS * EMB * E4 * 2);   // [L][E][4E]
  _Float16* WHt   = (_Float16*)carve((size_t)VOC * EMB * 2);           // [V][E]
  float*    XF    = (float*)   carve((size_t)TOK * EMB * 4);
  float*    PROJ  = (float*)   carve((size_t)TOK * EMB * 4);
  _Float16* XH    = (_Float16*)carve((size_t)TOK * EMB * 2);
  _Float16* QKV   = (_Float16*)carve((size_t)TOK * N3 * 2);   // contiguous w/ ATTN
  _Float16* ATTN  = (_Float16*)carve((size_t)TOK * EMB * 2);
  _Float16* VT    = (_Float16*)carve((size_t)TOK * EMB * 2);
  _Float16* FFH   = QKV;  // [TOK,E4] aliases dead QKV+ATTN (exact fit)

  // -------- one-time weight conversion to f16, K-innermost --------
  pack_qkv_kernel<<<2048, 256, 0, stream>>>(Wq, Wk, Wv, WQKV);
  cvt_t_kernel<<<2048, 256, 0, stream>>>(Wo,    WOt, LAYERS, EMB, EMB);
  cvt_t_kernel<<<2048, 256, 0, stream>>>(W1,    W1t, LAYERS, EMB, E4);
  cvt_t_kernel<<<2048, 256, 0, stream>>>(W2,    W2t, LAYERS, E4, EMB);
  cvt_t_kernel<<<1024, 256, 0, stream>>>(Whead, WHt, 1,      EMB, VOC);

  // -------- embedding --------
  embed_kernel<<<2048, 256, 0, stream>>>(indices, tok, pos, XF, XH);

  const dim3 blk(128);
  for (int l = 0; l < LAYERS; ++l) {
    // fused QKV projection: [TOK,E] x [3E,E]^T -> QKV f16
    gemm_wmma_f16<<<dim3(N3 / 64, TOK / 64), blk, 0, stream>>>(
        XH, WQKV + (size_t)l * N3 * EMB, nullptr, QKV, nullptr,
        TOK, N3, EMB, 0);
    // V slice -> seq-innermost VT
    repack_vt_kernel<<<2048, 256, 0, stream>>>(QKV, VT);
    // causal flash attention -> ATTN f16 [TOK,E]
    attn_flash<<<dim3(SEQ / 64, BATCH * NHEAD), blk, 0, stream>>>(QKV, VT, ATTN);
    // output projection + bias -> PROJ f32
    gemm_wmma_f16<<<dim3(EMB / 64, TOK / 64), blk, 0, stream>>>(
        ATTN, WOt + (size_t)l * EMB * EMB, PROJ, nullptr, bo + (size_t)l * EMB,
        TOK, EMB, EMB, 0);
    // x = LN(x + proj)
    ln_kernel<<<TOK, 256, 0, stream>>>(PROJ, XF, ln1g + (size_t)l * EMB,
                                       ln1b + (size_t)l * EMB, XF, XH);
    // FFN up + GELU -> FFH f16 [TOK,E4]
    gemm_wmma_f16<<<dim3(E4 / 64, TOK / 64), blk, 0, stream>>>(
        XH, W1t + (size_t)l * E4 * EMB, nullptr, FFH, b1 + (size_t)l * E4,
        TOK, E4, EMB, 1);
    // FFN down -> PROJ f32
    gemm_wmma_f16<<<dim3(EMB / 64, TOK / 64), blk, 0, stream>>>(
        FFH, W2t + (size_t)l * EMB * E4, PROJ, nullptr, b2 + (size_t)l * EMB,
        TOK, EMB, E4, 0);
    // x = LN(x + ff)
    ln_kernel<<<TOK, 256, 0, stream>>>(PROJ, XF, ln2g + (size_t)l * EMB,
                                       ln2b + (size_t)l * EMB, XF, XH);
  }

  // final LN (f16 copy only) + head GEMM -> f32 logits in d_out
  ln_kernel<<<TOK, 256, 0, stream>>>(XF, nullptr, lnfg, lnfb, nullptr, XH);
  gemm_wmma_f16<<<dim3(VOC / 64, TOK / 64), blk, 0, stream>>>(
      XH, WHt, (float*)d_out, nullptr, bhead, TOK, VOC, EMB, 0);
}
